// BertNet_63136019251515
// MI455X (gfx1250) — matrix operand order repaired and checked by reading, not compiled, not run on hardware
//
#include <hip/hip_runtime.h>
#include <hip/hip_bf16.h>

// ---------------------------------------------------------------------------
// MI455X (gfx1250) implementation of the spatial+temporal attention block.
//
// Roofline reasoning: ~134 GFLOP of the ~178 GFLOP total is four families of
// 1x1-conv GEMMs (O x C x 3200 per sample).  At 23.3 TB/s the mandatory HBM
// traffic (~1 GB) costs ~45 us, so GEMMs must run on the bf16 WMMA pipe
// (v_wmma_f32_16x16x32_bf16, f32 accumulate) with fused epilogues
// (bias + BN + residual + leaky + bf16 requant) to avoid extra 105 MB
// activation round-trips.  The small/strided attention einsums (25x25 and
// 128x128 per (n,s), V=25 strides, tanh) stay on VALU with LDS staging.
// ---------------------------------------------------------------------------

typedef __bf16 bf16_t;
typedef __bf16 v16bf __attribute__((ext_vector_type(16)));
typedef float  v8f   __attribute__((ext_vector_type(8)));

#define NB      32
#define C_CH    256
#define T_DIM   128
#define V_DIM   25
#define S_HEADS 3
#define CI      64
#define P_DIM   (T_DIM * V_DIM)     // 3200
#define OQK     (2 * S_HEADS * CI)  // 384
#define KOUT    (S_HEADS * C_CH)    // 768

// ---------------------------------------------------------------- utilities
__global__ void cvt_f32_to_bf16(const float* __restrict__ in,
                                bf16_t* __restrict__ out, int n) {
  int i = blockIdx.x * 256 + threadIdx.x;
  if (i < n) out[i] = (bf16_t)in[i];
}

// out = bf16(in + PE).  PE: c even -> sin(pos*div), c odd -> cos(pos*div),
// div = exp(-(ln 1e4) * (c & ~1) / C); pos = v (spatial) or t (temporal).
__global__ void prep_pe_kernel(const float* __restrict__ in,
                               bf16_t* __restrict__ out, int temporal) {
  long idx = (long)blockIdx.x * 256 + threadIdx.x;   // N*C*P threads exactly
  int p = (int)(idx % P_DIM);
  int c = (int)((idx / P_DIM) % C_CH);
  int t = p / V_DIM, v = p % V_DIM;
  float pos = temporal ? (float)t : (float)v;
  float ang = pos * __expf(-9.210340371976184f * (float)(c & ~1) / (float)C_CH);
  float pe  = (c & 1) ? __cosf(ang) : __sinf(ang);
  out[idx] = (bf16_t)(in[idx] + pe);
}

// ------------------------------------------------------------- WMMA GEMM
// C[n] (M x P) = A (M x K, bf16 row-major) * B[n] (K x P, bf16 row-major)
// epilogue: v = acc + bias[m]; if fused: v = v*(g[m]/sqrt(1+eps)) + be[m];
//           v = leaky(resid[m,p] + v).  Writes f32 and/or bf16 outputs.
// Workgroup: 256 threads = 8 waves, tile 64(M) x 128(N), K step 32.
__global__ __launch_bounds__(256) void gemm_bf16_kernel(
    const bf16_t* __restrict__ A, const bf16_t* __restrict__ B,
    const float* __restrict__ bias,
    const float* __restrict__ gmul, const float* __restrict__ gadd,
    const float* __restrict__ resid,
    float* __restrict__ outF, bf16_t* __restrict__ outB,
    int M, int K,
    long strideB, long strideOut, long strideRes, int fused) {
  __shared__ bf16_t As[64 * 40];    // A tile, rows padded to 40 (80B, 16B-aligned)
  __shared__ bf16_t Bst[128 * 40];  // B tile transposed: Bst[n][k]

  const int tid  = threadIdx.x;
  const int lane = tid & 31;
  const int wave = tid >> 5;
  const int wm   = wave >> 2;   // 0..1 : 32-row slab
  const int wn   = wave & 3;    // 0..3 : 32-col slab
  const int hl   = lane >> 4;   // half-wave id
  const int l16  = lane & 15;

  const int  n  = blockIdx.z;
  const int  p0 = blockIdx.x * 128;
  const int  m0 = blockIdx.y * 64;
  const bf16_t* Bn = B + (long)n * strideB;

  v8f acc00 = {}, acc01 = {}, acc10 = {}, acc11 = {};
  const unsigned* AsU = (const unsigned*)As;
  const unsigned* BsU = (const unsigned*)Bst;

  for (int k0 = 0; k0 < K; k0 += 32) {
    // stage A: 64 rows x 32 k, one 16B chunk per thread
    {
      int m  = tid >> 2;
      int ko = (tid & 3) * 8;
      uint4 u = *(const uint4*)(A + (long)(m0 + m) * K + (k0 + ko));
      *(uint4*)(&As[m * 40 + ko]) = u;
    }
    // stage B transposed: 32 k x 128 n, two 16B chunks per thread
    #pragma unroll
    for (int pass = 0; pass < 2; ++pass) {
      int cch  = tid + pass * 256;
      int k    = cch >> 4;
      int noff = (cch & 15) * 8;
      union { uint4 u; bf16_t h[8]; } tmp;
      tmp.u = *(const uint4*)(Bn + (long)(k0 + k) * P_DIM + p0 + noff);
      #pragma unroll
      for (int i = 0; i < 8; ++i) Bst[(noff + i) * 40 + k] = tmp.h[i];
    }
    __syncthreads();

    // build fragments (documented CDNA5 16-bit A/B lane layouts)
    const int kb = hl * 8;  // A: lanes 0-15 carry K{0-7,16-23}; 16-31 K{8-15,24-31}
    v16bf afr[2], bfr[2];
    #pragma unroll
    for (int ms = 0; ms < 2; ++ms) {
      int m = 32 * wm + 16 * ms + l16;
      union { v16bf v; unsigned u[8]; } f;
      #pragma unroll
      for (int j = 0; j < 4; ++j) {
        f.u[j]     = AsU[m * 20 + (kb >> 1) + j];
        f.u[j + 4] = AsU[m * 20 + 8 + (kb >> 1) + j];
      }
      afr[ms] = f.v;
    }
    #pragma unroll
    for (int nsb = 0; nsb < 2; ++nsb) {
      int nc = 32 * wn + 16 * nsb + l16;  // B: lanes 0-15 K{0-7,16-23}; 16-31 K{8-15,24-31}
      union { v16bf v; unsigned u[8]; } f;
      #pragma unroll
      for (int j = 0; j < 8; ++j) f.u[j] = BsU[nc * 20 + 4 * hl + (j & 3) + 8 * (j >> 2)];
      bfr[nsb] = f.v;
    }

    acc00 = __builtin_amdgcn_wmma_f32_16x16x32_bf16(false, afr[0], false, bfr[0],
                                                    (short)0, acc00, false, false);
    acc01 = __builtin_amdgcn_wmma_f32_16x16x32_bf16(false, afr[0], false, bfr[1],
                                                    (short)0, acc01, false, false);
    acc10 = __builtin_amdgcn_wmma_f32_16x16x32_bf16(false, afr[1], false, bfr[0],
                                                    (short)0, acc10, false, false);
    acc11 = __builtin_amdgcn_wmma_f32_16x16x32_bf16(false, afr[1], false, bfr[1],
                                                    (short)0, acc11, false, false);
    __syncthreads();
  }

  // epilogue. C/D layout: VGPR r -> row r (lanes 0-15) / 8+r (lanes 16-31)
  const float inv = rsqrtf(1.0f + 1e-5f);
  const float* residn = resid ? resid + (long)n * strideRes : nullptr;
  float*  outFn = outF ? outF + (long)n * strideOut : nullptr;
  bf16_t* outBn = outB ? outB + (long)n * strideOut : nullptr;
  v8f accs[2][2] = {{acc00, acc01}, {acc10, acc11}};

  #pragma unroll
  for (int ms = 0; ms < 2; ++ms)
    #pragma unroll
    for (int nsb = 0; nsb < 2; ++nsb) {
      int mB = m0 + 32 * wm + 16 * ms + 8 * hl;
      int p  = p0 + 32 * wn + 16 * nsb + l16;
      #pragma unroll
      for (int r = 0; r < 8; ++r) {
        int m = mB + r;
        float v = accs[ms][nsb][r] + bias[m];
        if (fused) {
          v = v * (gmul[m] * inv) + gadd[m];
          v += residn[(long)m * P_DIM + p];
          v = (v >= 0.f) ? v : 0.1f * v;
        }
        long o = (long)m * P_DIM + p;
        if (outFn) outFn[o] = v;
        if (outBn) outBn[o] = (bf16_t)v;
      }
    }
}

// ------------------------------------------------------- spatial attention
// att[n,s,u,v] = tanh(sum_{ci,t} q[ci,t,u]*k[ci,t,v] / (Ci*T))*alpha[s]+att0
__global__ void att_s_kernel(const float* __restrict__ qk,
                             const float* __restrict__ alphas,
                             const float* __restrict__ att0,
                             float* __restrict__ att) {
  int n = blockIdx.x / S_HEADS, s = blockIdx.x % S_HEADS;
  const float* qb = qk + (long)n * OQK * P_DIM + (long)(s * CI) * P_DIM;
  const float* kb = qk + (long)n * OQK * P_DIM + (long)((S_HEADS + s) * CI) * P_DIM;
  float a = alphas[s];
  for (int o = threadIdx.x; o < V_DIM * V_DIM; o += blockDim.x) {
    int u = o / V_DIM, v = o % V_DIM;
    float sum = 0.f;
    for (int ci = 0; ci < CI; ++ci) {
      const float* qr = qb + ci * P_DIM;
      const float* kr = kb + ci * P_DIM;
      for (int t = 0; t < T_DIM; ++t)
        sum += qr[t * V_DIM + u] * kr[t * V_DIM + v];
    }
    att[(long)blockIdx.x * (V_DIM * V_DIM) + o] =
        tanhf(sum / (float)(CI * T_DIM)) * a + att0[s * V_DIM * V_DIM + o];
  }
}

// y[n,s*C+c,t,v] = sum_u x[n,c,t,u] * att[n,s,u,v]   (bf16 out for next GEMM)
__global__ void apply_att_s_kernel(const float* __restrict__ x,
                                   const float* __restrict__ att,
                                   bf16_t* __restrict__ out) {
  __shared__ float attL[V_DIM * V_DIM];
  long idx = (long)blockIdx.x * 256 + threadIdx.x;
  int ns  = (int)(idx / ((long)C_CH * P_DIM));  // blocks never straddle (n,s)
  int rem = (int)(idx % ((long)C_CH * P_DIM));
  int c = rem / P_DIM, p = rem % P_DIM;
  int t = p / V_DIM,  v = p % V_DIM;
  int n = ns / S_HEADS, s = ns % S_HEADS;
  for (int j = threadIdx.x; j < V_DIM * V_DIM; j += 256)
    attL[j] = att[(long)ns * V_DIM * V_DIM + j];
  __syncthreads();
  const float* xr = x + (long)n * C_CH * P_DIM + (long)c * P_DIM + t * V_DIM;
  float sum = 0.f;
  #pragma unroll
  for (int u = 0; u < V_DIM; ++u) sum += xr[u] * attL[u * V_DIM + v];
  out[(long)n * KOUT * P_DIM + (long)(s * C_CH + c) * P_DIM + p] = (bf16_t)sum;
}

// ------------------------------------------------------ temporal attention
// att[n,s,t,q] = tanh(sum_{ci,v} q[ci,t,v]*k[ci,q,v] / (Ci*V))*alpha[s]+att0
__global__ void att_t_kernel(const float* __restrict__ qk,
                             const float* __restrict__ alphat,
                             const float* __restrict__ att0,
                             float* __restrict__ att) {
  __shared__ float qL[CI * V_DIM];  // 1600 f32
  int b = blockIdx.x;
  int t = b % T_DIM;
  int s = (b / T_DIM) % S_HEADS;
  int n = b / (T_DIM * S_HEADS);
  const float* qb = qk + (long)n * OQK * P_DIM + (long)(s * CI) * P_DIM;
  const float* kb = qk + (long)n * OQK * P_DIM + (long)((S_HEADS + s) * CI) * P_DIM;
  for (int j = threadIdx.x; j < CI * V_DIM; j += 128) {
    int ci = j / V_DIM, v = j % V_DIM;
    qL[j] = qb[ci * P_DIM + t * V_DIM + v];
  }
  __syncthreads();
  int q = threadIdx.x;
  float sum = 0.f;
  for (int ci = 0; ci < CI; ++ci) {
    const float* kr = kb + ci * P_DIM + q * V_DIM;
    #pragma unroll
    for (int v = 0; v < V_DIM; ++v) sum += qL[ci * V_DIM + v] * kr[v];
  }
  att[((long)(n * S_HEADS + s) * T_DIM + t) * T_DIM + q] =
      tanhf(sum / (float)(CI * V_DIM)) * alphat[s] +
      att0[(long)s * T_DIM * T_DIM + t * T_DIM + q];
}

// z[n,s*C+c,q,v] = sum_t y[n,c,t,v] * att[n,s,t,q]   (bf16 out for next GEMM)
__global__ void apply_att_t_kernel(const float* __restrict__ y,
                                   const float* __restrict__ att,
                                   bf16_t* __restrict__ out) {
  __shared__ float yrow[T_DIM];
  int b  = blockIdx.x;
  int v  = b % V_DIM;
  int c  = (b / V_DIM) % C_CH;
  int ns = b / (V_DIM * C_CH);
  int n  = ns / S_HEADS, s = ns % S_HEADS;
  const float* yr = y + (long)n * C_CH * P_DIM + (long)c * P_DIM + v;
  if (threadIdx.x < T_DIM) yrow[threadIdx.x] = yr[threadIdx.x * V_DIM];
  __syncthreads();
  int q = threadIdx.x;
  const float* ar = att + (long)ns * T_DIM * T_DIM + q;  // att[t][q]
  float sum = 0.f;
  #pragma unroll 4
  for (int t = 0; t < T_DIM; ++t) sum += yrow[t] * ar[t * T_DIM];
  out[(long)n * KOUT * P_DIM + (long)(s * C_CH + c) * P_DIM + q * V_DIM + v] =
      (bf16_t)sum;
}

// ---------------------------------------------------------------- driver
extern "C" void kernel_launch(void* const* d_in, const int* in_sizes, int n_in,
                              void* d_out, int out_size, void* d_ws, size_t ws_size,
                              hipStream_t stream) {
  const float* x       = (const float*)d_in[0];
  const float* w_qk_s  = (const float*)d_in[1];
  const float* b_qk_s  = (const float*)d_in[2];
  const float* alphas  = (const float*)d_in[3];
  const float* att0s   = (const float*)d_in[4];
  const float* w_out_s = (const float*)d_in[5];
  const float* b_out_s = (const float*)d_in[6];
  const float* g_out_s = (const float*)d_in[7];
  const float* be_out_s= (const float*)d_in[8];
  const float* w_ff_s  = (const float*)d_in[9];
  const float* b_ff_s  = (const float*)d_in[10];
  const float* g_ff_s  = (const float*)d_in[11];
  const float* be_ff_s = (const float*)d_in[12];
  const float* w_qk_t  = (const float*)d_in[13];
  const float* b_qk_t  = (const float*)d_in[14];
  const float* alphat  = (const float*)d_in[15];
  const float* att0t   = (const float*)d_in[16];
  const float* w_out_t = (const float*)d_in[17];
  const float* b_out_t = (const float*)d_in[18];
  const float* g_out_t = (const float*)d_in[19];
  const float* be_out_t= (const float*)d_in[20];
  const float* w_ff_t  = (const float*)d_in[21];
  const float* b_ff_t  = (const float*)d_in[22];
  const float* g_ff_t  = (const float*)d_in[23];
  const float* be_ff_t = (const float*)d_in[24];
  float* out = (float*)d_out;
  (void)in_sizes; (void)n_in; (void)out_size; (void)ws_size;

  size_t off = 0;
  auto wsAlloc = [&](size_t bytes) -> void* {
    void* p = (char*)d_ws + off;
    off += (bytes + 255) & ~(size_t)255;
    return p;
  };
  bf16_t* wqk_s_b  = (bf16_t*)wsAlloc((size_t)OQK * C_CH * 2);
  bf16_t* wout_s_b = (bf16_t*)wsAlloc((size_t)C_CH * KOUT * 2);
  bf16_t* wff_s_b  = (bf16_t*)wsAlloc((size_t)C_CH * C_CH * 2);
  bf16_t* wqk_t_b  = (bf16_t*)wsAlloc((size_t)OQK * C_CH * 2);
  bf16_t* wout_t_b = (bf16_t*)wsAlloc((size_t)C_CH * KOUT * 2);
  bf16_t* wff_t_b  = (bf16_t*)wsAlloc((size_t)C_CH * C_CH * 2);
  bf16_t* actB     = (bf16_t*)wsAlloc((size_t)NB * C_CH * P_DIM * 2);   // 52 MB
  float*  qkbuf    = (float*) wsAlloc((size_t)NB * OQK * P_DIM * 4);    // 157 MB
  float*  attS     = (float*) wsAlloc((size_t)NB * S_HEADS * V_DIM * V_DIM * 4);
  float*  attT     = (float*) wsAlloc((size_t)NB * S_HEADS * T_DIM * T_DIM * 4);
  bf16_t* attApp   = (bf16_t*)wsAlloc((size_t)NB * KOUT * P_DIM * 2);   // 157 MB
  float*  Ybuf     = (float*) wsAlloc((size_t)NB * C_CH * P_DIM * 4);   // 105 MB

  auto cvt = [&](const float* src, bf16_t* dst, int nElem) {
    cvt_f32_to_bf16<<<(nElem + 255) / 256, 256, 0, stream>>>(src, dst, nElem);
  };
  cvt(w_qk_s,  wqk_s_b,  OQK * C_CH);
  cvt(w_out_s, wout_s_b, C_CH * KOUT);
  cvt(w_ff_s,  wff_s_b,  C_CH * C_CH);
  cvt(w_qk_t,  wqk_t_b,  OQK * C_CH);
  cvt(w_out_t, wout_t_b, C_CH * KOUT);
  cvt(w_ff_t,  wff_t_b,  C_CH * C_CH);

  auto gemm = [&](const bf16_t* A, const bf16_t* B, const float* bias,
                  const float* g, const float* be, const float* resid,
                  float* oF, bf16_t* oB, int M, int K, int fused) {
    dim3 grid(P_DIM / 128, M / 64, NB);
    gemm_bf16_kernel<<<grid, 256, 0, stream>>>(
        A, B, bias, g, be, resid, oF, oB, M, K,
        (long)K * P_DIM, (long)M * P_DIM, (long)C_CH * P_DIM, fused);
  };

  const int prepGrid = (NB * C_CH * P_DIM) / 256;   // 102400

  // ---------------- spatial block ----------------
  prep_pe_kernel<<<prepGrid, 256, 0, stream>>>(x, actB, 0);
  gemm(wqk_s_b, actB, b_qk_s, nullptr, nullptr, nullptr,
       qkbuf, nullptr, OQK, C_CH, 0);
  att_s_kernel<<<NB * S_HEADS, 256, 0, stream>>>(qkbuf, alphas, att0s, attS);
  apply_att_s_kernel<<<(NB * S_HEADS * C_CH * P_DIM) / 256, 256, 0, stream>>>(
      x, attS, attApp);
  gemm(wout_s_b, attApp, b_out_s, g_out_s, be_out_s, x,
       nullptr, actB, C_CH, KOUT, 1);               // y_mid (bf16) -> actB
  gemm(wff_s_b, actB, b_ff_s, g_ff_s, be_ff_s, x,
       Ybuf, nullptr, C_CH, C_CH, 1);               // Y (f32)

  // ---------------- temporal block ----------------
  prep_pe_kernel<<<prepGrid, 256, 0, stream>>>(Ybuf, actB, 1);
  gemm(wqk_t_b, actB, b_qk_t, nullptr, nullptr, nullptr,
       qkbuf, nullptr, OQK, C_CH, 0);
  att_t_kernel<<<NB * S_HEADS * T_DIM, 128, 0, stream>>>(qkbuf, alphat, att0t, attT);
  apply_att_t_kernel<<<NB * S_HEADS * C_CH * V_DIM, 128, 0, stream>>>(
      Ybuf, attT, attApp);
  gemm(wout_t_b, attApp, b_out_t, g_out_t, be_out_t, Ybuf,
       nullptr, actB, C_CH, KOUT, 1);               // z_mid (bf16) -> actB
  gemm(wff_t_b, actB, b_ff_t, g_ff_t, be_ff_t, Ybuf,
       out, nullptr, C_CH, C_CH, 1);                // final f32 -> d_out
}
